// SpeechDecoderWrapper_66709432041874
// MI455X (gfx1250) — matrix operand order repaired and checked by reading, not compiled, not run on hardware
//
#include <hip/hip_runtime.h>
#include <hip/hip_bf16.h>

// ---------------------------------------------------------------------------
// MI455X (gfx1250) speech-decoder pipeline.
// All GEMM-shaped work goes through one bf16 WMMA kernel
// (v_wmma_f32_16x16x32_bf16, f32 accum): 128x64 block tile, 32x32 wave tile
// (4 WMMA / wave / K-step), double-buffered LDS. Weights are pre-transposed
// and converted to bf16 [N,K] once per call, so weight-tile staging is a pure
// 16B/lane copy via GLOBAL_LOAD_ASYNC_TO_LDS_B128 (ASYNCcnt-tracked).
// NOTE: all GEMM K dims in this network are multiples of 32 (no K-edge path).
// ---------------------------------------------------------------------------

typedef __bf16        v16bf   __attribute__((ext_vector_type(16)));
typedef float         v8f     __attribute__((ext_vector_type(8)));
typedef unsigned int  uint4e  __attribute__((ext_vector_type(4)));
typedef float         float4e __attribute__((ext_vector_type(4)));

union FragU { uint4e q[2]; v16bf v; };

#if __has_builtin(__builtin_amdgcn_global_load_async_to_lds_b128)
#define GEMM_HAS_ASYNC 1
typedef int v4i __attribute__((vector_size(16)));
typedef __attribute__((address_space(1))) v4i* as1v4p;   // global int4*
typedef __attribute__((address_space(3))) v4i* as3v4p;   // LDS    int4*
#endif

__device__ __forceinline__ void wait_async() {
#ifdef GEMM_HAS_ASYNC
#if __has_builtin(__builtin_amdgcn_s_wait_asynccnt)
    __builtin_amdgcn_s_wait_asynccnt(0);
#else
    asm volatile("s_wait_asynccnt 0x0" ::: "memory");
#endif
#endif
}

__device__ __forceinline__ unsigned short f2bf(float f) {
    unsigned int u = __builtin_bit_cast(unsigned int, f);
    u += 0x7FFFu + ((u >> 16) & 1u);          // round-to-nearest-even
    return (unsigned short)(u >> 16);
}
__device__ __forceinline__ unsigned int pack2(float lo, float hi) {
    return (unsigned int)f2bf(lo) | ((unsigned int)f2bf(hi) << 16);
}

// Stage a 128x32 fp32 tile (row stride ld) -> LDS bf16, 20-dword (80B) pitch.
__device__ __forceinline__ void stage_a128(
    const float* __restrict__ src, int ld, int rows, int K,
    int r0, int k0, unsigned int* dst, int tid)
{
    const int r  = tid >> 1;                   // 0..127
    const int kh = (tid & 1) * 16;             // 0 or 16
    const int gr = r0 + r;
    float4e a0 = {0.f,0.f,0.f,0.f}, a1 = a0, a2 = a0, a3 = a0;
    if (gr < rows) {                           // single per-thread branch
        const float* p = src + (size_t)gr * ld + k0 + kh;
        if (k0 + 64 <= K) __builtin_prefetch(p + 32, 0, 3);   // global_prefetch_b8
        a0 = ((const float4e*)p)[0];
        a1 = ((const float4e*)p)[1];
        a2 = ((const float4e*)p)[2];
        a3 = ((const float4e*)p)[3];
    }
    uint4e w0, w1;
    w0.x = pack2(a0.x, a0.y);  w0.y = pack2(a0.z, a0.w);
    w0.z = pack2(a1.x, a1.y);  w0.w = pack2(a1.z, a1.w);
    w1.x = pack2(a2.x, a2.y);  w1.y = pack2(a2.z, a2.w);
    w1.z = pack2(a3.x, a3.y);  w1.w = pack2(a3.z, a3.w);
    uint4e* d = (uint4e*)&dst[r * 20 + (kh >> 1)];
    d[0] = w0;
    d[1] = w1;
}

// Stage a 64x32 tile from fp32 row-major [N,K] (B^T layout) -> LDS bf16.
__device__ __forceinline__ void stage_b_nk(
    const float* __restrict__ src, int ld, int rows, int K,
    int r0, int k0, unsigned int* dst, int tid)
{
    const int r  = tid >> 2;                   // 0..63
    const int kq = (tid & 3) * 8;              // 0,8,16,24
    const int gr = r0 + r;
    float4e a0 = {0.f,0.f,0.f,0.f}, a1 = a0;
    if (gr < rows) {
        const float* p = src + (size_t)gr * ld + k0 + kq;
        if (k0 + 64 <= K) __builtin_prefetch(p + 32, 0, 3);
        a0 = ((const float4e*)p)[0];
        a1 = ((const float4e*)p)[1];
    }
    uint4e w;
    w.x = pack2(a0.x, a0.y);  w.y = pack2(a0.z, a0.w);
    w.z = pack2(a1.x, a1.y);  w.w = pack2(a1.z, a1.w);
    *(uint4e*)&dst[r * 20 + (kq >> 1)] = w;
}

// Stage fp32 B[K,N] (row stride ldb) transposed into LDS dst[n][k] (bf16).
__device__ __forceinline__ void stage_b_kxn(
    const float* __restrict__ B, int ldb, int N, int K,
    int c0, int k0, unsigned int* dst, int tid)
{
    const int n  = tid & 63;
    const int gn = c0 + n;
    const int kb = (tid >> 6) * 8;             // 0,8,16,24
    float v[8] = {0.f,0.f,0.f,0.f,0.f,0.f,0.f,0.f};
    if (gn < N) {
        const float* p = B + (size_t)(k0 + kb) * ldb + gn;
        #pragma unroll
        for (int j = 0; j < 8; ++j) v[j] = p[(size_t)j * ldb];   // coalesced over n
    }
    uint4e w;
    w.x = pack2(v[0], v[1]);  w.y = pack2(v[2], v[3]);
    w.z = pack2(v[4], v[5]);  w.w = pack2(v[6], v[7]);
    *(uint4e*)&dst[n * 20 + (kb >> 1)] = w;
}

// Stage bf16 [N,K] weight tile -> LDS: pure 16B/lane copy (async if available).
__device__ __forceinline__ void stage_b_bf16(
    const unsigned short* __restrict__ src, int ldb, int N, int K,
    int c0, int k0, unsigned int* dst, int tid)
{
    const int r  = tid >> 2;                   // 0..63 (row n)
    const int q  = (tid & 3) * 8;              // bf16 elems: 0,8,16,24
    const int gn = c0 + r;
    unsigned int* lp = &dst[r * 20 + (q >> 1)];
    if (gn < N) {
        const unsigned short* gp = src + (size_t)gn * ldb + k0 + q;
        if (k0 + 64 <= K) __builtin_prefetch(gp + 32, 0, 3);
#ifdef GEMM_HAS_ASYNC
        __builtin_amdgcn_global_load_async_to_lds_b128(
            (as1v4p)(unsigned long long)(const void*)gp,
            (as3v4p)(unsigned int)(unsigned long long)(void*)lp, 0, 0);
#else
        *(uint4e*)lp = *(const uint4e*)gp;
#endif
    } else {
        uint4e z = {0u, 0u, 0u, 0u};
        *(uint4e*)lp = z;                      // N-edge rows -> zeros
    }
}

// ---------------------------------------------------------------------------
// Generic tiled GEMM: C[M,N] = epilogue(alpha * A[M,K] @ B + bias)
//   A: fp32 row-major, row stride lda (lda may differ from K -> im2col views)
//   B: tB==0 fp32 [K,N] ; tB==1 fp32 [N,K] ; tB==2 bf16 [N,K]
//   epilogue: act(0 none,1 GELU,2 ELU) -> *lsw[n] -> +res[m,n] -> +=C (accum)
// Requires K % 32 == 0.
// ---------------------------------------------------------------------------
__global__ __launch_bounds__(256) void gemm_bf16_k(
    const float* __restrict__ A, int lda,
    const void* __restrict__ B, int ldb, int tB,
    float* C, int ldc,
    int M, int N, int K,
    const float* __restrict__ bias, float alpha, int accum,
    const float* res, int ldres,
    const float* __restrict__ lsw, int act)
{
    __shared__ unsigned int AsAll[2 * 128 * 20];   // 2 x 10240 B
    __shared__ unsigned int BsAll[2 *  64 * 20];   // 2 x  5120 B

    const int tid  = threadIdx.x;
    const int row0 = blockIdx.x * 128;
    const int col0 = blockIdx.y * 64;
    const int lane = tid & 31;
    const int wv   = tid >> 5;                 // 8 waves
    const int wr   = wv >> 1;                  // wave row 0..3 (32 rows each)
    const int wc   = wv & 1;                   // wave col 0..1 (32 cols each)
    const int hf   = lane >> 4;
    const int l16  = lane & 15;

    v8f acc00 = {0.f,0.f,0.f,0.f,0.f,0.f,0.f,0.f};
    v8f acc01 = acc00, acc10 = acc00, acc11 = acc00;

    const int nk = K >> 5;

    auto stage = [&](int k0, int buf) {
        stage_a128(A, lda, M, K, row0, k0, &AsAll[buf * 2560], tid);
        unsigned int* bs = &BsAll[buf * 1280];
        if      (tB == 2) stage_b_bf16((const unsigned short*)B, ldb, N, K, col0, k0, bs, tid);
        else if (tB == 1) stage_b_nk  ((const float*)B, ldb, N, K, col0, k0, bs, tid);
        else              stage_b_kxn ((const float*)B, ldb, N, K, col0, k0, bs, tid);
    };

    stage(0, 0);
    for (int it = 0; it < nk; ++it) {
        wait_async();                          // drain async B staging (if any)
        __syncthreads();
        if (it + 1 < nk) stage((it + 1) << 5, (it + 1) & 1);   // overlap next tile

        const unsigned int* Ab = &AsAll[(it & 1) * 2560];
        const unsigned int* Bb = &BsAll[(it & 1) * 1280];

        FragU fa0, fa1, fb0, fb1;
        const int ar0 = wr * 32 + l16;         // A: two 8-elem K runs per lane
        fa0.q[0] = *(const uint4e*)&Ab[ar0 * 20 + hf * 4];
        fa0.q[1] = *(const uint4e*)&Ab[ar0 * 20 + 8 + hf * 4];
        const int ar1 = ar0 + 16;
        fa1.q[0] = *(const uint4e*)&Ab[ar1 * 20 + hf * 4];
        fa1.q[1] = *(const uint4e*)&Ab[ar1 * 20 + 8 + hf * 4];
        const int bn0 = wc * 32 + l16;         // B: two 8-elem K runs per lane
        fb0.q[0] = *(const uint4e*)&Bb[bn0 * 20 + hf * 4];
        fb0.q[1] = *(const uint4e*)&Bb[bn0 * 20 + 8 + hf * 4];
        const int bn1 = bn0 + 16;
        fb1.q[0] = *(const uint4e*)&Bb[bn1 * 20 + hf * 4];
        fb1.q[1] = *(const uint4e*)&Bb[bn1 * 20 + 8 + hf * 4];

        acc00 = __builtin_amdgcn_wmma_f32_16x16x32_bf16(false, fa0.v, false, fb0.v,
                                                        (short)0, acc00, false, false);
        acc01 = __builtin_amdgcn_wmma_f32_16x16x32_bf16(false, fa0.v, false, fb1.v,
                                                        (short)0, acc01, false, false);
        acc10 = __builtin_amdgcn_wmma_f32_16x16x32_bf16(false, fa1.v, false, fb0.v,
                                                        (short)0, acc10, false, false);
        acc11 = __builtin_amdgcn_wmma_f32_16x16x32_bf16(false, fa1.v, false, fb1.v,
                                                        (short)0, acc11, false, false);
    }

    // ---- epilogue: C layout m = v + hf*8 (+16*i), n = l16 (+16*j) ----
    #pragma unroll
    for (int i = 0; i < 2; ++i) {
        #pragma unroll
        for (int j = 0; j < 2; ++j) {
            v8f& acc = i ? (j ? acc11 : acc10) : (j ? acc01 : acc00);
            const int n = col0 + wc * 32 + j * 16 + l16;
            #pragma unroll
            for (int v = 0; v < 8; ++v) {
                const int m = row0 + wr * 32 + i * 16 + hf * 8 + v;
                if (m < M && n < N) {
                    float val = alpha * acc[v];
                    if (bias) val += bias[n];
                    if (act == 1)      val = 0.5f * val * (1.f + erff(val * 0.70710678118f));
                    else if (act == 2) val = (val > 0.f) ? val : expm1f(val);
                    if (lsw) val *= lsw[n];
                    if (res) val += res[(size_t)m * ldres + n];
                    const size_t ci = (size_t)m * ldc + n;
                    float o = val;
                    if (accum) o += C[ci];
                    C[ci] = o;
                }
            }
        }
    }
}

// ---------------------------------------------------------------------------
// Elementwise kernels
// ---------------------------------------------------------------------------
// fp32 [K,N] -> bf16 [N,K] transpose+convert (one-time weight prep)
__global__ void tcvt_k(const float* __restrict__ src, unsigned short* __restrict__ dst,
                       int K, int N)
{
    const long long idx = (long long)blockIdx.x * 256 + threadIdx.x;
    if (idx >= (long long)K * N) return;
    const int k = (int)(idx / N);
    const int n = (int)(idx - (long long)k * N);
    dst[(size_t)n * K + k] = f2bf(src[idx]);
}

__global__ void rmsnorm_k(const float* __restrict__ in, const float* __restrict__ w,
                          float* __restrict__ out, int D)
{
    const int row = blockIdx.x;
    const float* p = in  + (size_t)row * D;
    float*       q = out + (size_t)row * D;
    __shared__ float sh[256];
    float s = 0.f;
    for (int i = threadIdx.x; i < D; i += 256) { float v = p[i]; s += v * v; }
    sh[threadIdx.x] = s; __syncthreads();
    for (int o = 128; o > 0; o >>= 1) {
        if (threadIdx.x < o) sh[threadIdx.x] += sh[threadIdx.x + o];
        __syncthreads();
    }
    const float r = rsqrtf(sh[0] / (float)D + 1e-6f);
    for (int i = threadIdx.x; i < D; i += 256) q[i] = p[i] * r * w[i];
}

// qk-RMSNorm (over HD) + RoPE, in place on [T, H*HD].  grid=(T,H), block=HD(=64)
__global__ void qkrope_k(float* __restrict__ x, const float* __restrict__ w,
                         int Dtot, int HD)
{
    const int t  = blockIdx.x;
    const int hh = blockIdx.y;
    const int i  = threadIdx.x;
    float* p = x + (size_t)t * Dtot + hh * HD;
    const float v = p[i];
    __shared__ float sh[64];
    sh[i] = v * v; __syncthreads();
    for (int o = 32; o > 0; o >>= 1) {
        if (i < o) sh[i] += sh[i + o];
        __syncthreads();
    }
    const float r = rsqrtf(sh[0] / (float)HD + 1e-6f);
    const int hhd = HD >> 1;
    const int oi  = (i < hhd) ? i + hhd : i - hhd;
    const float ov  = p[oi] * r * w[oi];
    const float y   = v * r * w[i];
    const float rot = (i < hhd) ? -ov : ov;
    const int   fi  = (i < hhd) ? i : i - hhd;
    const float freq = __powf(10000.f, (-2.f * (float)fi) / (float)HD);
    const float ang  = (float)t * freq;
    const float outv = y * __cosf(ang) + rot * __sinf(ang);
    __syncthreads();                 // all reads of p done before any write
    p[i] = outv;
}

// causal softmax over S[h][row][*].  grid=(T,H), block=256
__global__ void softmax_k(float* __restrict__ S, int T)
{
    const int row = blockIdx.x, hh = blockIdx.y;
    float* p = S + ((size_t)hh * T + row) * T;
    const int n = row + 1;
    __shared__ float sh[256];
    float m = -1e30f;
    for (int i = threadIdx.x; i < n; i += 256) m = fmaxf(m, p[i]);
    sh[threadIdx.x] = m; __syncthreads();
    for (int o = 128; o > 0; o >>= 1) {
        if (threadIdx.x < o) sh[threadIdx.x] = fmaxf(sh[threadIdx.x], sh[threadIdx.x + o]);
        __syncthreads();
    }
    m = sh[0]; __syncthreads();
    float s = 0.f;
    for (int i = threadIdx.x; i < n; i += 256) { float e = __expf(p[i] - m); p[i] = e; s += e; }
    sh[threadIdx.x] = s; __syncthreads();
    for (int o = 128; o > 0; o >>= 1) {
        if (threadIdx.x < o) sh[threadIdx.x] += sh[threadIdx.x + o];
        __syncthreads();
    }
    const float inv = 1.f / sh[0];
    for (int i = threadIdx.x; i < T; i += 256) p[i] = (i < n) ? p[i] * inv : 0.f;
}

// RVQ codebook gather: out[t, :] = emb[codes[t], :]
__global__ void gather_k(const int* __restrict__ codes, const float* __restrict__ emb,
                         float* __restrict__ out, int T, int C)
{
    const int idx = blockIdx.x * 256 + threadIdx.x;
    if (idx >= T * C) return;
    const int t = idx / C, c = idx - t * C;
    out[idx] = emb[(size_t)codes[t] * C + c];
}

// x[t,c] += sum_n b[n,c]
__global__ void addprb_k(float* __restrict__ x, const float* __restrict__ b,
                         int T, int C, int Nq)
{
    const int idx = blockIdx.x * 256 + threadIdx.x;
    if (idx >= T * C) return;
    const int c = idx % C;
    float s = 0.f;
    for (int n = 0; n < Nq; ++n) s += b[n * C + c];
    x[idx] += s;
}

// pad / zero-stuff (transposed conv) with optional fused ELU on real samples
__global__ void stuff_k(const float* __restrict__ in, float* __restrict__ out,
                        long long total, int C, int Lin, int padL, int s, int elu)
{
    const long long idx = (long long)blockIdx.x * 256 + threadIdx.x;
    if (idx >= total) return;
    const long long t = idx / C;
    const int c = (int)(idx - t * C);
    const long long tt = t - padL;
    float v = 0.f;
    if (tt >= 0 && (tt % s) == 0) {
        const long long ti = tt / s;
        if (ti < Lin) {
            v = in[ti * C + c];
            if (elu && v < 0.f) v = expm1f(v);
        }
    }
    out[idx] = v;
}

// final conv: k=7 SAME, 32 -> 1, ELU on input, clip to [-1,1]
__global__ void finconv_k(const float* __restrict__ y, const float* __restrict__ w,
                          const float* __restrict__ b, float* __restrict__ out, int S)
{
    const int si = blockIdx.x * 256 + threadIdx.x;
    if (si >= S) return;
    float acc = b[0];
    #pragma unroll
    for (int j = 0; j < 7; ++j) {
        const int t = si - 3 + j;
        if (t >= 0 && t < S) {
            const float* row = y + (size_t)t * 32;
            #pragma unroll
            for (int c = 0; c < 32; ++c) {
                float v = row[c];
                if (v < 0.f) v = expm1f(v);
                acc += v * w[j * 32 + c];
            }
        }
    }
    out[si] = fminf(fmaxf(acc, -1.f), 1.f);
}

// ---------------------------------------------------------------------------
// Host orchestration
// ---------------------------------------------------------------------------
static inline void gemm(hipStream_t st, const float* A, int lda, const void* B, int ldb,
                        int tB, float* C, int ldc, int M, int N, int K,
                        const float* bias, float alpha, int accum,
                        const float* res, int ldres, const float* lsw, int act)
{
    dim3 gd((M + 127) / 128, (N + 63) / 64);
    gemm_bf16_k<<<gd, 256, 0, st>>>(A, lda, B, ldb, tB, C, ldc, M, N, K,
                                    bias, alpha, accum, res, ldres, lsw, act);
}

extern "C" void kernel_launch(void* const* d_in, const int* in_sizes, int n_in,
                              void* d_out, int out_size, void* d_ws, size_t ws_size,
                              hipStream_t stream)
{
    (void)in_sizes; (void)n_in; (void)out_size; (void)ws_size;

    const int T = 512, D = 512, H = 8, HD = 64, FF = 2048;
    const int LAT = 1024, CB = 2048, CBD = 256, L = 8;

    int a = 0;
    const int*   codes     = (const int*)d_in[a++];
    const float* emb_first = (const float*)d_in[a++];
    const float* pf_w  = (const float*)d_in[a++];
    const float* pf_b  = (const float*)d_in[a++];
    const float* fo_w  = (const float*)d_in[a++];
    const float* fo_b  = (const float*)d_in[a++];
    const float* emb_rest = (const float*)d_in[a++];
    const float* pr_w  = (const float*)d_in[a++];
    const float* pr_b  = (const float*)d_in[a++];
    const float* ro_w  = (const float*)d_in[a++];
    const float* ro_b  = (const float*)d_in[a++];
    const float* pre_w = (const float*)d_in[a++];
    const float* pre_b = (const float*)d_in[a++];
    const float* in_w  = (const float*)d_in[a++];
    const float* in_b  = (const float*)d_in[a++];
    const float* ln1   = (const float*)d_in[a++];
    const float* qw    = (const float*)d_in[a++];
    const float* kw    = (const float*)d_in[a++];
    const float* vw    = (const float*)d_in[a++];
    const float* ow    = (const float*)d_in[a++];
    const float* qn    = (const float*)d_in[a++];
    const float* kn    = (const float*)d_in[a++];
    const float* ls_a  = (const float*)d_in[a++];
    const float* ln2   = (const float*)d_in[a++];
    const float* w1    = (const float*)d_in[a++];
    const float* w2    = (const float*)d_in[a++];
    const float* ls_m  = (const float*)d_in[a++];
    const float* fn_w  = (const float*)d_in[a++];
    const float* out_w = (const float*)d_in[a++];
    const float* out_b = (const float*)d_in[a++];
    const float* up_w  = (const float*)d_in[a++];
    const float* up_b  = (const float*)d_in[a++];
    const float* dec0_w = (const float*)d_in[a++];
    const float* dec0_b = (const float*)d_in[a++];
    const float* dw1 = (const float*)d_in[a++];
    const float* db1 = (const float*)d_in[a++];
    const float* dw2 = (const float*)d_in[a++];
    const float* db2 = (const float*)d_in[a++];
    const float* dw3 = (const float*)d_in[a++];
    const float* db3 = (const float*)d_in[a++];
    const float* dw4 = (const float*)d_in[a++];
    const float* db4 = (const float*)d_in[a++];
    const float* fin_w = (const float*)d_in[a++];
    const float* fin_b = (const float*)d_in[a++];

    // ---- workspace bump allocator: fp32 region then bf16 weight arena ----
    size_t off = 0;
    float* W = (float*)d_ws;
    auto F = [&](size_t n) { float* p = W + off; off += n; return p; };

    float* gbuf   = F((size_t)T * CBD);
    float* tmp    = F((size_t)T * LAT);
    float* qf     = F((size_t)T * LAT);
    float* qr     = F((size_t)T * LAT);
    float* hidden = F((size_t)T * LAT);
    float* pad1   = F((size_t)(T + 4) * LAT);
    float* hid2   = F((size_t)T * LAT);
    float* hb     = F((size_t)T * D);
    float* xb     = F((size_t)T * D);
    float* Qb     = F((size_t)T * D);
    float* Kb     = F((size_t)T * D);
    float* Vb     = F((size_t)T * D);
    float* Ob     = F((size_t)T * D);
    float* Sc     = F((size_t)H * T * T);
    float* t1     = F((size_t)T * FF);
    float* g      = F((size_t)T * LAT);
    float* ustuff = F((size_t)1028 * LAT);
    float* u      = F((size_t)1024 * LAT);
    float* Rstuff = F((size_t)983048 * 64);     // largest zero-stuffed buffer
    float* Rya    = F((size_t)983040 * 32);     // ping
    float* Ryb    = F((size_t)983040 * 32);     // pong

    unsigned short* WB = (unsigned short*)(W + off);
    size_t boff = 0;
    auto FB = [&](size_t n) { unsigned short* p = WB + boff; boff += n; return p; };

    auto blocks1d = [](long long n) { return (unsigned)((n + 255) / 256); };
    auto tc = [&](const float* src, unsigned short* dst, int K, int N) {
        tcvt_k<<<blocks1d((long long)K * N), 256, 0, stream>>>(src, dst, K, N);
    };

    // ---- one-time weight transpose+convert to bf16 [N,K] ----
    unsigned short* pfT  = FB((size_t)LAT * CBD);        tc(pf_w, pfT, CBD, LAT);
    unsigned short* foT  = FB((size_t)LAT * LAT);        tc(fo_w, foT, LAT, LAT);
    unsigned short* prT  = FB((size_t)15 * LAT * CBD);
    for (int n = 0; n < 15; ++n)
        tc(pr_w + (size_t)n * CBD * LAT, prT + (size_t)n * LAT * CBD, CBD, LAT);
    unsigned short* roT  = FB((size_t)LAT * LAT);        tc(ro_w, roT, LAT, LAT);
    unsigned short* preT = FB((size_t)LAT * 3 * LAT);    tc(pre_w, preT, 3 * LAT, LAT);
    unsigned short* inT  = FB((size_t)D * LAT);          tc(in_w, inT, LAT, D);
    unsigned short* qwT  = FB((size_t)L * D * D);
    unsigned short* kwT  = FB((size_t)L * D * D);
    unsigned short* vwT  = FB((size_t)L * D * D);
    unsigned short* owT  = FB((size_t)L * D * D);
    unsigned short* w1T  = FB((size_t)L * FF * D);
    unsigned short* w2T  = FB((size_t)L * D * FF);
    for (int l = 0; l < L; ++l) {
        tc(qw + (size_t)l * D * D, qwT + (size_t)l * D * D, D, D);
        tc(kw + (size_t)l * D * D, kwT + (size_t)l * D * D, D, D);
        tc(vw + (size_t)l * D * D, vwT + (size_t)l * D * D, D, D);
        tc(ow + (size_t)l * D * D, owT + (size_t)l * D * D, D, D);
        tc(w1 + (size_t)l * D * FF, w1T + (size_t)l * FF * D, D, FF);
        tc(w2 + (size_t)l * FF * D, w2T + (size_t)l * D * FF, FF, D);
    }
    unsigned short* outT  = FB((size_t)LAT * D);         tc(out_w, outT, D, LAT);
    unsigned short* upT   = FB((size_t)LAT * 4 * LAT);   tc(up_w, upT, 4 * LAT, LAT);
    unsigned short* dec0T = FB((size_t)512 * 7 * LAT);   tc(dec0_w, dec0T, 7 * LAT, 512);
    const int   Ks[4]  = {16, 12, 10, 8};
    const int   ss[4]  = {8, 6, 5, 4};
    const int   Ci[4]  = {512, 256, 128, 64};
    const int   Co[4]  = {256, 128, 64, 32};
    const float* dwv[4] = {dw1, dw2, dw3, dw4};
    const float* dbv[4] = {db1, db2, db3, db4};
    unsigned short* dwT[4];
    for (int i = 0; i < 4; ++i) {
        dwT[i] = FB((size_t)Co[i] * Ks[i] * Ci[i]);
        tc(dwv[i], dwT[i], Ks[i] * Ci[i], Co[i]);
    }

    // ================= RVQ decode =================
    gather_k<<<blocks1d((long long)T * CBD), 256, 0, stream>>>(codes, emb_first, gbuf, T, CBD);
    gemm(stream, gbuf, CBD, pfT, CBD, 2, tmp, LAT, T, LAT, CBD, pf_b, 1.f, 0, nullptr, 0, nullptr, 0);
    gemm(stream, tmp, LAT, foT, LAT, 2, qf, LAT, T, LAT, LAT, fo_b, 1.f, 0, nullptr, 0, nullptr, 0);
    for (int n = 0; n < 15; ++n) {
        gather_k<<<blocks1d((long long)T * CBD), 256, 0, stream>>>(
            codes + (n + 1) * T, emb_rest + (size_t)n * CB * CBD, gbuf, T, CBD);
        gemm(stream, gbuf, CBD, prT + (size_t)n * LAT * CBD, CBD, 2, qr, LAT,
             T, LAT, CBD, nullptr, 1.f, n > 0, nullptr, 0, nullptr, 0);
    }
    addprb_k<<<blocks1d((long long)T * LAT), 256, 0, stream>>>(qr, pr_b, T, LAT, 15);
    gemm(stream, qr, LAT, roT, LAT, 2, hidden, LAT, T, LAT, LAT, ro_b, 1.f, 0, qf, LAT, nullptr, 0);

    // ============ causal pre-conv (k=3, left pad 2) via im2col GEMM ============
    stuff_k<<<blocks1d((long long)(T + 2) * LAT), 256, 0, stream>>>(
        hidden, pad1, (long long)(T + 2) * LAT, LAT, T, 2, 1, 0);
    gemm(stream, pad1, LAT, preT, 3 * LAT, 2, hid2, LAT, T, LAT, 3 * LAT, pre_b, 1.f, 0, nullptr, 0, nullptr, 0);

    // ============ in_proj ============
    gemm(stream, hid2, LAT, inT, LAT, 2, hb, D, T, D, LAT, in_b, 1.f, 0, nullptr, 0, nullptr, 0);

    // ============ transformer (L layers) ============
    const float scale = 0.125f;   // HD^-0.5
    for (int l = 0; l < L; ++l) {
        rmsnorm_k<<<T, 256, 0, stream>>>(hb, ln1 + l * D, xb, D);
        gemm(stream, xb, D, qwT + (size_t)l * D * D, D, 2, Qb, D, T, D, D, nullptr, 1.f, 0, nullptr, 0, nullptr, 0);
        gemm(stream, xb, D, kwT + (size_t)l * D * D, D, 2, Kb, D, T, D, D, nullptr, 1.f, 0, nullptr, 0, nullptr, 0);
        gemm(stream, xb, D, vwT + (size_t)l * D * D, D, 2, Vb, D, T, D, D, nullptr, 1.f, 0, nullptr, 0, nullptr, 0);

        qkrope_k<<<dim3(T, H), HD, 0, stream>>>(Qb, qn + l * HD, D, HD);
        qkrope_k<<<dim3(T, H), HD, 0, stream>>>(Kb, kn + l * HD, D, HD);

        for (int h = 0; h < H; ++h)   // scores = scale * Q_h @ K_h^T (fp32 [N,K] path)
            gemm(stream, Qb + h * HD, D, Kb + h * HD, D, 1, Sc + (size_t)h * T * T, T,
                 T, T, HD, nullptr, scale, 0, nullptr, 0, nullptr, 0);
        softmax_k<<<dim3(T, H), 256, 0, stream>>>(Sc, T);
        for (int h = 0; h < H; ++h)   // O_h = P_h @ V_h (fp32 [K,N] path)
            gemm(stream, Sc + (size_t)h * T * T, T, Vb + h * HD, D, 0, Ob + h * HD, D,
                 T, HD, T, nullptr, 1.f, 0, nullptr, 0, nullptr, 0);

        gemm(stream, Ob, D, owT + (size_t)l * D * D, D, 2, hb, D, T, D, D,
             nullptr, 1.f, 0, hb, D, ls_a + l * D, 0);

        rmsnorm_k<<<T, 256, 0, stream>>>(hb, ln2 + l * D, xb, D);
        gemm(stream, xb, D, w1T + (size_t)l * FF * D, D, 2, t1, FF, T, FF, D,
             nullptr, 1.f, 0, nullptr, 0, nullptr, 1);
        gemm(stream, t1, FF, w2T + (size_t)l * D * FF, FF, 2, hb, D, T, D, FF,
             nullptr, 1.f, 0, hb, D, ls_m + l * D, 0);
    }

    // ============ final norm + out proj ============
    rmsnorm_k<<<T, 256, 0, stream>>>(hb, fn_w, xb, D);
    gemm(stream, xb, D, outT, D, 2, g, LAT, T, LAT, D, out_b, 1.f, 0, nullptr, 0, nullptr, 0);

    // ============ upsample x2: convT(k=4,s=2) + ELU ============
    stuff_k<<<blocks1d((long long)1028 * LAT), 256, 0, stream>>>(
        g, ustuff, (long long)1028 * LAT, LAT, T, 1, 2, 0);
    gemm(stream, ustuff, LAT, upT, 4 * LAT, 2, u, LAT, 1024, LAT, 4 * LAT, up_b, 1.f, 0, nullptr, 0, nullptr, 2);

    // ============ dec0: conv k=7 SAME ============
    stuff_k<<<blocks1d((long long)1030 * LAT), 256, 0, stream>>>(
        u, Rstuff, (long long)1030 * LAT, LAT, 1024, 3, 1, 0);
    float* cur = Rya;
    gemm(stream, Rstuff, LAT, dec0T, 7 * LAT, 2, cur, 512, 1024, 512, 7 * LAT, dec0_b, 1.f, 0, nullptr, 0, nullptr, 0);

    // ============ 4x [ELU -> convT] (SEANet upsampling stack) ============
    int Lin = 1024;
    for (int i = 0; i < 4; ++i) {
        const int Lout = Lin * ss[i];
        const long long Lst = (long long)Lout + Ks[i];
        const int padL = (Ks[i] - ss[i]) / 2;
        stuff_k<<<blocks1d(Lst * Ci[i]), 256, 0, stream>>>(
            cur, Rstuff, Lst * Ci[i], Ci[i], Lin, padL, ss[i], 1 /*fused ELU*/);
        float* nxt = (i & 1) ? Rya : Ryb;
        gemm(stream, Rstuff, Ci[i], dwT[i], Ks[i] * Ci[i], 2, nxt, Co[i],
             Lout, Co[i], Ks[i] * Ci[i], dbv[i], 1.f, 0, nullptr, 0, nullptr, 0);
        cur = nxt;
        Lin = Lout;
    }

    // ============ final conv k=7 SAME, 32->1, ELU + clip ============
    finconv_k<<<blocks1d(983040), 256, 0, stream>>>(cur, fin_w, fin_b, (float*)d_out, 983040);
}
